// PointnetBIG_74345883894236
// MI455X (gfx1250) — compile-verified, matrix operand order checked by reading, not run on hardware
//
#include <hip/hip_runtime.h>
#include <hip/hip_bf16.h>

typedef _Float16 h16 __attribute__((ext_vector_type(16)));
typedef _Float16 h8  __attribute__((ext_vector_type(8)));
typedef float    f8  __attribute__((ext_vector_type(8)));

#define NPTS   16384
#define NPART  32
#define FDIM   1024
#define TILE   64      // points per workgroup
#define NWAVE  8       // 256 threads / wave32

// Layer geometry (cin padded to multiple of 32 for WMMA K)
constexpr int CIN  [8] = {  3, 64, 64, 64,128,256,512, 512};
constexpr int CINP [8] = { 32, 64, 64, 64,128,256,512, 512};
constexpr int COUT [8] = { 64, 64, 64,128,256,512,512,1024};
// halves offset of each layer's fragment-layout weights in ws
constexpr int WOFF [8] = {0, 2048, 6144, 10240, 18432, 51200, 182272, 444416};
// float offset of each layer's folded bias in ws
constexpr int BOFF [8] = {0, 64, 128, 192, 320, 576, 1088, 1600};

// LDS layout (bytes): bufA [0,66560) | bufB [66560,133120) | seg ids [133120,133376)
#define LDS_BUF_BYTES 66560
#define LDS_TOTAL     133376

// ---------------------------------------------------------------------------
// Fold BN into weights/bias; emit weights in per-lane WMMA B-fragment layout:
// fragment ft = nt*ktiles + kt; flat = (ft*32 + lane)*16 + h
// lane<16: n = nt*16+lane,     k = kt*32 + (h<8 ? h : 8+h)
// lane>=16: n = nt*16+lane-16, k = kt*32 + (h<8 ? h : 8+h) + 8
// ---------------------------------------------------------------------------
__global__ void prep_layer_kernel(const float* __restrict__ w,
                                  const float* __restrict__ b,
                                  const float* __restrict__ gamma,
                                  const float* __restrict__ beta,
                                  const float* __restrict__ mean,
                                  const float* __restrict__ var,
                                  int cin, int cinp, int cout,
                                  _Float16* __restrict__ wdst,
                                  float* __restrict__ bdst) {
    int tid = blockIdx.x * blockDim.x + threadIdx.x;
    int total = cout * cinp;
    if (tid < total) {
        int h     = tid & 15;
        int lane  = (tid >> 4) & 31;
        int ft    = tid >> 9;
        int ktiles = cinp >> 5;
        int nt = ft / ktiles;
        int kt = ft - nt * ktiles;
        int n  = nt * 16 + (lane & 15);
        int kl = (h < 8 ? h : 8 + h) + ((lane >= 16) ? 8 : 0);
        int k  = kt * 32 + kl;
        float val = 0.0f;
        if (k < cin && n < cout) {
            float sc = gamma[n] * rsqrtf(var[n] + 1e-5f);
            val = w[n * cin + k] * sc;
        }
        wdst[tid] = (_Float16)val;
    }
    if (tid < cout) {
        float sc = gamma[tid] * rsqrtf(var[tid] + 1e-5f);
        bdst[tid] = (b[tid] - mean[tid]) * sc + beta[tid];
    }
}

__global__ void zero_out_kernel(float* __restrict__ p, int n) {
    int i = blockIdx.x * blockDim.x + threadIdx.x;
    if (i < n) p[i] = 0.0f;
}

// ---------------------------------------------------------------------------
// One fused layer: out[64, cout] = relu( in[64, cin] * W^T + b' ), all in LDS.
// Each wave owns PAIRS of adjacent n-tiles (two independent accumulators
// sharing one A fragment -> back-to-back independent v_wmma, half the LDS
// A traffic, no hazard NOPs).  Tile pair t -> (mt = t&3, nt0 = 2*(t>>2)).
// LI == 7 pools straight to global memory via u32 atomicMax (values >= 0).
// ---------------------------------------------------------------------------
template <int LI>
__device__ __forceinline__ void layer_mm(const _Float16* __restrict__ inBuf,
                                         _Float16* __restrict__ outBuf,
                                         const _Float16* __restrict__ wfrag,
                                         const float* __restrict__ biasBase,
                                         const int* __restrict__ sSeg,
                                         unsigned int* __restrict__ outG,
                                         int wave, int lane) {
    constexpr int cinp   = CINP[LI];
    constexpr int cout   = COUT[LI];
    constexpr int sIn    = cinp + 8;      // halves; keeps 16B alignment
    constexpr int sOut   = cout + 8;
    constexpr int ktiles = cinp / 32;
    constexpr int ntiles = cout / 16;     // always even
    constexpr int tpairs = 4 * (ntiles / 2);

    const _Float16* wf = wfrag + WOFF[LI];
    const float*    bs = biasBase + BOFF[LI];

    const int lx     = lane & 15;
    const int hiHalf = lane >> 4;         // 0 or 1

    for (int t = wave; t < tpairs; t += NWAVE) {
        int mt  = t & 3;
        int nt0 = 2 * (t >> 2);
        int nt1 = nt0 + 1;

        float bv0 = bs[nt0 * 16 + lx];
        float bv1 = bs[nt1 * 16 + lx];
        f8 acc0 = {bv0, bv0, bv0, bv0, bv0, bv0, bv0, bv0};
        f8 acc1 = {bv1, bv1, bv1, bv1, bv1, bv1, bv1, bv1};

        const int prowA = mt * 16 + lx;   // A-matrix row for this lane
        for (int kt = 0; kt < ktiles; ++kt) {
            // A fragment: two contiguous 16B runs per lane from LDS (shared)
            const _Float16* ap = inBuf + prowA * sIn + kt * 32 + hiHalf * 8;
            h8 alo = *(const h8*)ap;
            h8 ahi = *(const h8*)(ap + 16);
            h16 a;
#pragma unroll
            for (int i = 0; i < 8; ++i) { a[i] = alo[i]; a[8 + i] = ahi[i]; }

            // B fragments: one 32B contiguous load per lane per n-tile
            h16 b0 = *(const h16*)(wf + (size_t)((nt0 * ktiles + kt) * 32 + lane) * 16);
            h16 b1 = *(const h16*)(wf + (size_t)((nt1 * ktiles + kt) * 32 + lane) * 16);

            acc0 = __builtin_amdgcn_wmma_f32_16x16x32_f16(
                false, a, false, b0, (short)0, acc0, false, false);
            acc1 = __builtin_amdgcn_wmma_f32_16x16x32_f16(
                false, a, false, b1, (short)0, acc1, false, false);
        }

        const int ch0  = nt0 * 16 + lx;
        const int ch1  = nt1 * 16 + lx;
        const int prow = mt * 16 + hiHalf * 8;
        if (LI < 7) {
#pragma unroll
            for (int r = 0; r < 8; ++r) {
                _Float16* row = outBuf + (prow + r) * sOut;
                row[ch0] = (_Float16)__builtin_fmaxf(acc0[r], 0.0f);  // fused ReLU
                row[ch1] = (_Float16)__builtin_fmaxf(acc1[r], 0.0f);
            }
        } else {
#pragma unroll
            for (int r = 0; r < 8; ++r) {
                float v0 = __builtin_fmaxf(acc0[r], 0.0f);
                float v1 = __builtin_fmaxf(acc1[r], 0.0f);
                int seg = sSeg[prow + r];
                unsigned int* segrow = outG + (size_t)seg * FDIM;
                atomicMax(segrow + ch0, __float_as_uint(v0));
                atomicMax(segrow + ch1, __float_as_uint(v1));
            }
        }
    }
}

__global__ void __launch_bounds__(256)
pointnet_fused_kernel(const float* __restrict__ x,
                      const int* __restrict__ pid,
                      const _Float16* __restrict__ wfrag,
                      const float* __restrict__ bias,
                      unsigned int* __restrict__ outG) {
    extern __shared__ char smem[];
    _Float16* bufA = (_Float16*)smem;
    _Float16* bufB = (_Float16*)(smem + LDS_BUF_BYTES);
    int*      sSeg = (int*)(smem + 2 * LDS_BUF_BYTES);

    const int tid   = threadIdx.x;
    const int wave  = tid >> 5;
    const int lane  = tid & 31;
    const int gbase = blockIdx.x * TILE;

    // Stage layer-0 input tile: (point, ch) with ch padded 3 -> 32 (zeros)
    for (int i = tid; i < TILE * 32; i += blockDim.x) {
        int p = i >> 5;
        int c = i & 31;
        int g = gbase + p;
        int b = g >> 14;            // /16384
        int n = g & (NPTS - 1);
        float v = (c < 3) ? x[((size_t)b * 3 + c) * NPTS + n] : 0.0f;
        bufA[p * (CINP[0] + 8) + c] = (_Float16)v;
    }
    if (tid < TILE) {
        int g = gbase + tid;
        int b = g >> 14;
        sSeg[tid] = b * NPART + pid[g];
    }
    __syncthreads();

    layer_mm<0>(bufA, bufB, wfrag, bias, sSeg, outG, wave, lane); __syncthreads();
    layer_mm<1>(bufB, bufA, wfrag, bias, sSeg, outG, wave, lane); __syncthreads();
    layer_mm<2>(bufA, bufB, wfrag, bias, sSeg, outG, wave, lane); __syncthreads();
    layer_mm<3>(bufB, bufA, wfrag, bias, sSeg, outG, wave, lane); __syncthreads();
    layer_mm<4>(bufA, bufB, wfrag, bias, sSeg, outG, wave, lane); __syncthreads();
    layer_mm<5>(bufB, bufA, wfrag, bias, sSeg, outG, wave, lane); __syncthreads();
    layer_mm<6>(bufA, bufB, wfrag, bias, sSeg, outG, wave, lane); __syncthreads();
    layer_mm<7>(bufB, bufA, wfrag, bias, sSeg, outG, wave, lane);
}

extern "C" void kernel_launch(void* const* d_in, const int* in_sizes, int n_in,
                              void* d_out, int out_size, void* d_ws, size_t ws_size,
                              hipStream_t stream) {
    const float* x   = (const float*)d_in[0];
    const int*   pid = (const int*)d_in[1];
    // d_in[2] is P (scalar, == 32, hardcoded)

    float*    bias  = (float*)d_ws;                       // 2624 floats
    _Float16* wfrag = (_Float16*)((char*)d_ws + 16384);   // 968704 halves

    // 1) zero the pooled output (max identity for post-ReLU values)
    {
        int n = out_size;
        zero_out_kernel<<<(n + 255) / 256, 256, 0, stream>>>((float*)d_out, n);
    }

    // 2) fold BN into weights/bias, swizzle weights to WMMA fragment layout
    for (int li = 0; li < 8; ++li) {
        const float* w  = (const float*)d_in[3 + 6 * li + 0];
        const float* b  = (const float*)d_in[3 + 6 * li + 1];
        const float* g  = (const float*)d_in[3 + 6 * li + 2];
        const float* bt = (const float*)d_in[3 + 6 * li + 3];
        const float* mn = (const float*)d_in[3 + 6 * li + 4];
        const float* vr = (const float*)d_in[3 + 6 * li + 5];
        int total = COUT[li] * CINP[li];
        int work  = total > COUT[li] ? total : COUT[li];
        prep_layer_kernel<<<(work + 255) / 256, 256, 0, stream>>>(
            w, b, g, bt, mn, vr, CIN[li], CINP[li], COUT[li],
            wfrag + WOFF[li], bias + BOFF[li]);
    }

    // 3) fused 8-layer MLP + segment-max pool
    const int nblocks = (8 * NPTS) / TILE;   // 2048
    pointnet_fused_kernel<<<nblocks, 256, LDS_TOTAL, stream>>>(
        x, pid, wfrag, bias, (unsigned int*)d_out);
}
